// TGNLayerGraphAttentionEmbedding_21174188769661
// MI455X (gfx1250) — matrix operand order repaired and checked by reading, not compiled, not run on hardware
//
#include <hip/hip_runtime.h>
#include <math.h>

// ---------------------------------------------------------------------------
// TGN graph-attention layer for MI455X (gfx1250, wave32, WMMA + TDM).
// Memory-bound problem (~57 GFLOP, ~150MB traffic @ 23.3 TB/s): bf16 staging
// of all GEMM operands (working set ~80MB fits in 192MB L2),
// v_wmma_f32_16x16x32_bf16 for every matmul, streaming (flash) softmax so the
// 2x4096x4096 score tensor never exists in memory, and Tensor-Data-Mover
// double-buffered A-tile staging (tensor_load_to_lds / s_wait_tensorcnt) for
// the two large 6144-deep GEMMs.
// ---------------------------------------------------------------------------

typedef __attribute__((ext_vector_type(16))) __bf16        v16bf;
typedef __attribute__((ext_vector_type(8)))  float         v8f;
typedef __attribute__((ext_vector_type(4)))  unsigned int  v4u;
typedef __attribute__((ext_vector_type(8)))  int           v8i;
typedef __attribute__((ext_vector_type(4)))  int           v4i;

union FragU { v16bf f; v4u q[2]; };

constexpr int N_NODES = 4096;
constexpr int KNB     = 16;
constexpr int EMB     = 256;
constexpr int EDGE    = 64;
constexpr int TIMEF   = 64;
constexpr int QDIM    = EMB + TIMEF;        // 320
constexpr int KD      = EMB + EDGE + TIMEF; // 384
constexpr int AGGW    = KNB * KD;           // 6144
constexpr int HEADS   = 2;
constexpr int DH      = QDIM / HEADS;       // 160
constexpr int XDIM    = QDIM + EMB;         // 576

#define DEV __device__ __forceinline__

DEV unsigned short f2bf(float x) {
  union { float f; unsigned u; } c; c.f = x;
  unsigned r = c.u + 0x7FFFu + ((c.u >> 16) & 1u);   // round-to-nearest-even
  return (unsigned short)(r >> 16);
}

// A-fragment (16x32 bf16, M x K): lane l -> row l&15; halves 0..7 = K base+0..7,
// halves 8..15 = K base+16..23, where base = kOff + 8*(l>>4).  (ISA 7.12.2)
DEV v16bf load_fragA(const unsigned short* base, int ld, int kOff, int lane) {
  int row = lane & 15;
  int hb  = (lane >> 4) << 3;
  const unsigned short* p = base + (size_t)row * ld + kOff + hb;
  FragU u;
  u.q[0] = *(const v4u*)(p);
  u.q[1] = *(const v4u*)(p + 16);
  return u.f;
}

// B-fragment (32x16 bf16, K x N) for C = A * B with B = W^T (W row-major
// [Nout, Kdim]): lane l -> column l&15 (a row of W); halves 0..7 = K base+0..7,
// halves 8..15 = K base+16..23, where base = kOff + 8*(l>>4).
DEV v16bf load_fragB(const unsigned short* base, int ld, int kOff, int lane) {
  int col = lane & 15;
  int kb  = (lane >> 4) << 3;
  const unsigned short* p = base + (size_t)col * ld + kOff + kb;
  FragU u;
  u.q[0] = *(const v4u*)(p);
  u.q[1] = *(const v4u*)(p + 16);
  return u.f;
}

// ---------------------------------------------------------------------------
// TDM: DMA a 16-row x 32-col bf16 tile (row stride = ld elements) from global
// memory into LDS.  D# descriptor per CDNA5 ISA:
//   group0: count=1 | lds_addr | 57-bit global_addr | type=2
//   group1: data_size=1 (2B), tensor_dim0=ld, tensor_dim1=rows,
//           tile_dim0=32, tile_dim1=16, tensor_dim0_stride=ld
//   groups 2/3 (and the trailing group): zero (2-D tensor).
// Per-wave issue; completion tracked with TENSORcnt (per-wave), so no
// cross-wave synchronization is required.
// ---------------------------------------------------------------------------
DEV void tdm_load_tile16x32(const unsigned short* gsrc, unsigned lds_byte_addr,
                            int ld) {
  unsigned long long ga = (unsigned long long)(size_t)gsrc;
  v4u g0;
  g0[0] = 1u;                                            // count=1, user mode
  g0[1] = lds_byte_addr;                                 // LDS byte address
  g0[2] = (unsigned)(ga & 0xffffffffu);                  // global addr [31:0]
  g0[3] = (unsigned)((ga >> 32) & 0x01ffffffu)           // global addr [56:32]
          | (2u << 30);                                  // type = 2 ("image")
  v8i g1;
  g1[0] = 1 << 16;                                       // data_size=1 (2B)
  g1[1] = (int)((ld & 0xffff) << 16);                    // tensor_dim0 lo16
  g1[2] = (int)(((unsigned)ld >> 16) & 0xffffu)          // tensor_dim0 hi16
          | (16 << 16);                                  // tensor_dim1 = 16
  g1[3] = (int)(32u << 16);                              // tile_dim0 = 32
  g1[4] = 16;                                            // tile_dim1 = 16
  g1[5] = ld;                                            // dim0_stride lo32
  g1[6] = 0;                                             // dim0_stride hi16
  g1[7] = 0;
  v4i gz4 = {0, 0, 0, 0};
  v8i gz8 = {0, 0, 0, 0, 0, 0, 0, 0};
  // amdgpu-toolchain (clang-23) 6-arg form:
  // (uint32x4 g0, int32x8 g1, int32x4 g2, int32x4 g3, int32x8, i32 cpol)
  __builtin_amdgcn_tensor_load_to_lds(g0, g1, gz4, gz4, gz8, 0);
}

// ---------------------------------------------------------------------------
// Generic bf16 WMMA GEMM:  out[M,Nout] = act( (A[M,K] @ W[Nout,K]^T + bias) * scale )
// Block: 256 threads = 8 waves; wave computes a 16(M) x 64(N) strip.
// grid.x = M/128, grid.y = Nout/64.  All dims divide evenly here.
// USE_TDM: A-tiles staged through wave-private LDS by the Tensor Data Mover,
// double-buffered (issue tile k+1 while WMMA consumes tile k).
// ---------------------------------------------------------------------------
template<bool RELU, bool OUT_BF16, bool TRANS_OUT, bool USE_TDM>
__global__ __launch_bounds__(256)
void gemm_bf16_wmma(const unsigned short* __restrict__ A, int lda,
                    const unsigned short* __restrict__ W, int ldw,
                    const float* __restrict__ bias, float scale,
                    void* __restrict__ out, int ldc, int Kdim) {
  const int lane  = threadIdx.x & 31;
  const int wave  = threadIdx.x >> 5;
  const int mtile = blockIdx.x * 128 + wave * 16;
  const int ntile = blockIdx.y * 64;

  v8f acc[4];
#pragma unroll
  for (int t = 0; t < 4; ++t)
#pragma unroll
    for (int r = 0; r < 8; ++r) acc[t][r] = 0.f;

  const unsigned short* Abase = A + (size_t)mtile * lda;

  auto mm_step = [&](v16bf a, int kb) {
#pragma unroll
    for (int t = 0; t < 4; ++t) {
      v16bf b = load_fragB(W + (size_t)(ntile + t * 16) * ldw, ldw, kb, lane);
      acc[t] = __builtin_amdgcn_wmma_f32_16x16x32_bf16(
          false, a, false, b, (short)0, acc[t], false, false);
    }
  };

  if constexpr (USE_TDM) {
    // 8 waves x 2 buffers x (16x32) bf16 = 16 KB of the 320KB WGP LDS.
    __shared__ __align__(16) unsigned short atile[8][2][16 * 32];
    unsigned short* buf[2] = { &atile[wave][0][0], &atile[wave][1][0] };
    // Flat->LDS offset: LDS aperture tag lives in addr[63:32]; addr[31:0] is
    // the LDS byte offset (ISA 10.2 aperture mapping).
    unsigned lds_off[2] = { (unsigned)(size_t)buf[0], (unsigned)(size_t)buf[1] };

    tdm_load_tile16x32(Abase, lds_off[0], lda);
    int cur = 0;
    int kb = 0;
    for (; kb < Kdim - 32; kb += 32, cur ^= 1) {
      tdm_load_tile16x32(Abase + kb + 32, lds_off[cur ^ 1], lda);
      __builtin_amdgcn_s_wait_tensorcnt(1);     // previous tile landed
      asm volatile("" ::: "memory");
      mm_step(load_fragA(buf[cur], 32, 0, lane), kb);
    }
    __builtin_amdgcn_s_wait_tensorcnt(0);       // final tile landed
    asm volatile("" ::: "memory");
    mm_step(load_fragA(buf[cur], 32, 0, lane), kb);
  } else {
    for (int kb = 0; kb < Kdim; kb += 32) {
      if (kb + 64 < Kdim)
        __builtin_prefetch(Abase + (size_t)(lane & 15) * lda + kb + 64, 0, 0);
      mm_step(load_fragA(Abase, lda, kb, lane), kb);
    }
  }

  // C layout: lane holds column n = lane&15; VGPR r -> row r + 8*(lane>>4).
  const int n  = lane & 15;
  const int hi = lane >> 4;
#pragma unroll
  for (int t = 0; t < 4; ++t) {
    int col = ntile + t * 16 + n;
    float bv = bias ? bias[col] : 0.f;
#pragma unroll
    for (int r = 0; r < 8; ++r) {
      int row = mtile + r + hi * 8;
      float v = (acc[t][r] + bv) * scale;
      if (RELU) v = fmaxf(v, 0.f);
      if (TRANS_OUT)
        ((unsigned short*)out)[(size_t)col * ldc + row] = f2bf(v);
      else if (OUT_BF16)
        ((unsigned short*)out)[(size_t)row * ldc + col] = f2bf(v);
      else
        ((float*)out)[(size_t)row * ldc + col] = v;
    }
  }
}

// ---------------------------------------------------------------------------
// Flash attention, one head per grid.y, 128 query rows per block (16/wave).
// q is pre-scaled by 1/sqrt(dh).  k: [N,320] row-major bf16; vT: [320,N] bf16.
// P (16x32) is bounced through wave-private LDS to convert C-layout -> A-frag.
// ---------------------------------------------------------------------------
__global__ __launch_bounds__(256)
void flash_attn_wmma(const unsigned short* __restrict__ qbf,
                     const unsigned short* __restrict__ kbf,
                     const unsigned short* __restrict__ vT,
                     unsigned short* __restrict__ obf) {
  __shared__ __align__(16) unsigned short plds[8][16 * 32];

  const int lane = threadIdx.x & 31;
  const int wave = threadIdx.x >> 5;
  const int hoff = blockIdx.y * DH;
  const int q0   = blockIdx.x * 128 + wave * 16;
  const int n    = lane & 15;
  const int hi   = lane >> 4;

  v16bf qf[5];                                   // dh = 160 = 5 * 32
#pragma unroll
  for (int kk = 0; kk < 5; ++kk)
    qf[kk] = load_fragA(qbf + (size_t)q0 * QDIM + hoff, QDIM, kk * 32, lane);

  v8f oacc[10];                                  // 16 x 160 output accumulator
#pragma unroll
  for (int j = 0; j < 10; ++j)
#pragma unroll
    for (int r = 0; r < 8; ++r) oacc[j][r] = 0.f;

  float mrow[8], lrow[8];
#pragma unroll
  for (int r = 0; r < 8; ++r) { mrow[r] = -INFINITY; lrow[r] = 0.f; }

  unsigned short* myp = plds[wave];

  for (int kb = 0; kb < N_NODES; kb += 32) {
    // S = Q @ K^T for 32 keys (two 16x16 C-tiles)
    v8f s[2];
#pragma unroll
    for (int t = 0; t < 2; ++t) {
      v8f sa;
#pragma unroll
      for (int r = 0; r < 8; ++r) sa[r] = 0.f;
#pragma unroll
      for (int kk = 0; kk < 5; ++kk) {
        v16bf bk = load_fragB(kbf + (size_t)(kb + t * 16) * QDIM + hoff,
                              QDIM, kk * 32, lane);
        sa = __builtin_amdgcn_wmma_f32_16x16x32_bf16(
            false, qf[kk], false, bk, (short)0, sa, false, false);
      }
      s[t] = sa;
    }

    // streaming softmax; row stats reduced across each 16-lane half
#pragma unroll
    for (int r = 0; r < 8; ++r) {
      float x = fmaxf(s[0][r], s[1][r]);
      x = fmaxf(x, __shfl_xor(x, 1));
      x = fmaxf(x, __shfl_xor(x, 2));
      x = fmaxf(x, __shfl_xor(x, 4));
      x = fmaxf(x, __shfl_xor(x, 8));
      float mnew = fmaxf(mrow[r], x);
      float c    = __expf(mrow[r] - mnew);       // 0 on first block (-inf)
      float p0   = __expf(s[0][r] - mnew);
      float p1   = __expf(s[1][r] - mnew);
      float rs = p0 + p1;
      rs += __shfl_xor(rs, 1);
      rs += __shfl_xor(rs, 2);
      rs += __shfl_xor(rs, 4);
      rs += __shfl_xor(rs, 8);
      lrow[r] = lrow[r] * c + rs;
      mrow[r] = mnew;
#pragma unroll
      for (int j = 0; j < 10; ++j) oacc[j][r] *= c;
      int prow = r + hi * 8;
      myp[prow * 32 + n]      = f2bf(p0);
      myp[prow * 32 + 16 + n] = f2bf(p1);
    }

    // O += P @ V ; P reloaded from LDS in A-fragment layout (DS in-order per wave)
    v16bf pf = load_fragA(myp, 32, 0, lane);
#pragma unroll
    for (int j = 0; j < 10; ++j) {
      v16bf vf = load_fragB(vT + (size_t)(hoff + j * 16) * N_NODES,
                            N_NODES, kb, lane);
      oacc[j] = __builtin_amdgcn_wmma_f32_16x16x32_bf16(
          false, pf, false, vf, (short)0, oacc[j], false, false);
    }
  }

#pragma unroll
  for (int r = 0; r < 8; ++r) {
    float inv = 1.f / lrow[r];
    int row = q0 + r + hi * 8;
#pragma unroll
    for (int j = 0; j < 10; ++j)
      obf[(size_t)row * QDIM + hoff + j * 16 + n] = f2bf(oacc[j][r] * inv);
  }
}

// ---------------------------------------------------------------------------
// Staging kernels: gather/concat -> bf16
// ---------------------------------------------------------------------------
__global__ void pack_agg_kernel(const float* __restrict__ feat,
                                const int* __restrict__ neigh,
                                const float* __restrict__ ef,
                                const float* __restrict__ tf,
                                unsigned short* __restrict__ agg) {
  const long total = (long)N_NODES * KNB * KD;
  for (long i = blockIdx.x * (long)blockDim.x + threadIdx.x; i < total;
       i += (long)gridDim.x * blockDim.x) {
    int  c    = (int)(i % KD);
    long rk   = i / KD;
    int  j    = (int)(rk % KNB);
    int  node = (int)(rk / KNB);
    float v;
    if (c < EMB)             v = feat[(size_t)neigh[node * KNB + j] * EMB + c];
    else if (c < EMB + EDGE) v = ef[((size_t)node * KNB + j) * EDGE + (c - EMB)];
    else                     v = tf[((size_t)node * KNB + j) * TIMEF + (c - EMB - EDGE)];
    agg[i] = f2bf(v);
  }
}

__global__ void pack_query_kernel(const float* __restrict__ feat,
                                  const float* __restrict__ tz,
                                  unsigned short* __restrict__ qb) {
  const long total = (long)N_NODES * QDIM;
  for (long i = blockIdx.x * (long)blockDim.x + threadIdx.x; i < total;
       i += (long)gridDim.x * blockDim.x) {
    int c = (int)(i % QDIM);
    int node = (int)(i / QDIM);
    float v = (c < EMB) ? feat[(size_t)node * EMB + c] : tz[c - EMB];
    qb[i] = f2bf(v);
  }
}

__global__ void pack_xfeat_kernel(const float* __restrict__ feat,
                                  unsigned short* __restrict__ xb) {
  const long total = (long)N_NODES * EMB;
  for (long i = blockIdx.x * (long)blockDim.x + threadIdx.x; i < total;
       i += (long)gridDim.x * blockDim.x) {
    int c = (int)(i % EMB);
    int node = (int)(i / EMB);
    xb[(size_t)node * XDIM + c] = f2bf(feat[i]);
  }
}

__global__ void conv_bf16_kernel(const float* __restrict__ src,
                                 unsigned short* __restrict__ dst, long cnt) {
  for (long i = blockIdx.x * (long)blockDim.x + threadIdx.x; i < cnt;
       i += (long)gridDim.x * blockDim.x)
    dst[i] = f2bf(src[i]);
}

// ---------------------------------------------------------------------------
extern "C" void kernel_launch(void* const* d_in, const int* in_sizes, int n_in,
                              void* d_out, int out_size, void* d_ws, size_t ws_size,
                              hipStream_t stream) {
  (void)in_sizes; (void)n_in; (void)out_size; (void)ws_size;
  const float* features  = (const float*)d_in[0];
  const int*   neighbors = (const int*)  d_in[1];
  const float* edgef     = (const float*)d_in[2];
  const float* timef     = (const float*)d_in[3];
  const float* tzeros    = (const float*)d_in[4];
  const float* Wq = (const float*)d_in[5],  *bq = (const float*)d_in[6];
  const float* Wk = (const float*)d_in[7],  *bk = (const float*)d_in[8];
  const float* Wv = (const float*)d_in[9],  *bv = (const float*)d_in[10];
  const float* Wo = (const float*)d_in[11], *bo = (const float*)d_in[12];
  const float* W1 = (const float*)d_in[13], *b1 = (const float*)d_in[14];
  const float* W2 = (const float*)d_in[15], *b2 = (const float*)d_in[16];
  float* out = (float*)d_out;

  char* ws = (char*)d_ws;
  size_t off = 0;
  auto alloc = [&](size_t elems) -> unsigned short* {
    unsigned short* p = (unsigned short*)(ws + off);
    off = (off + elems * 2 + 255) & ~(size_t)255;
    return p;
  };
  unsigned short* agg_bf   = alloc((size_t)N_NODES * AGGW);  // ~50 MB
  unsigned short* query_bf = alloc((size_t)N_NODES * QDIM);
  unsigned short* q_bf     = alloc((size_t)N_NODES * QDIM);
  unsigned short* k_bf     = alloc((size_t)N_NODES * QDIM);
  unsigned short* vT_bf    = alloc((size_t)QDIM * N_NODES);
  unsigned short* o_bf     = alloc((size_t)N_NODES * QDIM);
  unsigned short* x_bf     = alloc((size_t)N_NODES * XDIM);
  unsigned short* h_bf     = alloc((size_t)N_NODES * EMB);
  unsigned short* wq_bf    = alloc((size_t)QDIM * QDIM);
  unsigned short* wk_bf    = alloc((size_t)QDIM * AGGW);
  unsigned short* wv_bf    = alloc((size_t)QDIM * AGGW);
  unsigned short* wo_bf    = alloc((size_t)QDIM * QDIM);
  unsigned short* w1_bf    = alloc((size_t)EMB * XDIM);
  unsigned short* w2_bf    = alloc((size_t)EMB * EMB);

  // Stage weights + activations as bf16 (everything then lives in L2).
  conv_bf16_kernel<<<256, 256, 0, stream>>>(Wq, wq_bf, (long)QDIM * QDIM);
  conv_bf16_kernel<<<512, 256, 0, stream>>>(Wk, wk_bf, (long)QDIM * AGGW);
  conv_bf16_kernel<<<512, 256, 0, stream>>>(Wv, wv_bf, (long)QDIM * AGGW);
  conv_bf16_kernel<<<256, 256, 0, stream>>>(Wo, wo_bf, (long)QDIM * QDIM);
  conv_bf16_kernel<<<256, 256, 0, stream>>>(W1, w1_bf, (long)EMB * XDIM);
  conv_bf16_kernel<<<256, 256, 0, stream>>>(W2, w2_bf, (long)EMB * EMB);
  pack_agg_kernel<<<2048, 256, 0, stream>>>(features, neighbors, edgef, timef, agg_bf);
  pack_query_kernel<<<512, 256, 0, stream>>>(features, tzeros, query_bf);
  pack_xfeat_kernel<<<512, 256, 0, stream>>>(features, x_bf);

  const float qscale = 1.0f / sqrtf((float)DH);
  // q = (query @ Wq^T + bq) * 1/sqrt(dh)          -> bf16 [N,320]
  gemm_bf16_wmma<false, true, false, false><<<dim3(32, 5), 256, 0, stream>>>(
      query_bf, QDIM, wq_bf, QDIM, bq, qscale, q_bf, QDIM, QDIM);
  // k = agg @ Wk^T + bk   (TDM-staged A tiles)    -> bf16 [N,320]
  gemm_bf16_wmma<false, true, false, true><<<dim3(32, 5), 256, 0, stream>>>(
      agg_bf, AGGW, wk_bf, AGGW, bk, 1.f, k_bf, QDIM, AGGW);
  // v = agg @ Wv^T + bv, stored TRANSPOSED (TDM)  -> bf16 [320,N]
  gemm_bf16_wmma<false, true, true, true><<<dim3(32, 5), 256, 0, stream>>>(
      agg_bf, AGGW, wv_bf, AGGW, bv, 1.f, vT_bf, N_NODES, AGGW);
  // streaming 2-head attention                    -> bf16 [N,320]
  flash_attn_wmma<<<dim3(N_NODES / 128, HEADS), 256, 0, stream>>>(
      q_bf, k_bf, vT_bf, o_bf);
  // attn_out = o @ Wo^T + bo, written into x[:,256:576]
  gemm_bf16_wmma<false, true, false, false><<<dim3(32, 5), 256, 0, stream>>>(
      o_bf, QDIM, wo_bf, QDIM, bo, 1.f, x_bf + EMB, XDIM, QDIM);
  // h = relu(x @ W1^T + b1)                       -> bf16 [N,256]
  gemm_bf16_wmma<true, true, false, false><<<dim3(32, 4), 256, 0, stream>>>(
      x_bf, XDIM, w1_bf, XDIM, b1, 1.f, h_bf, EMB, XDIM);
  // out = h @ W2^T + b2                           -> f32 [N,256]
  gemm_bf16_wmma<false, false, false, false><<<dim3(32, 4), 256, 0, stream>>>(
      h_bf, EMB, w2_bf, EMB, b2, 1.f, out, EMB, EMB);
}
